// NN_LSTM_46634754900236
// MI455X (gfx1250) — compile-verified, matrix operand order checked by reading, not run on hardware
//
#include <hip/hip_runtime.h>
#include <hip/hip_bf16.h>
#include <math.h>

#define N_PED 6144
#define HID 256
#define OUT_DIM 32
#define NTOP 4
#define KCAT 288            // 32 (emb) + 256 (h)
#define GATES 1024          // 4*HID
#define ROWS 32             // rows per block in fused kernel

typedef __attribute__((ext_vector_type(8)))  _Float16 v8h_t;
typedef __attribute__((ext_vector_type(16))) _Float16 v16h_t;
typedef __attribute__((ext_vector_type(8)))  float    v8f_t;

// ---------------------------------------------------------------------------
// WMMA operand loader (A 16x32 or B 32x16, f16).
// rowbase = &mat[(tile_base + (lane&15)) * stride]; per-lane layout:
//   lanes 0-15 : K = kc*32 + {0..7}  and kc*32 + {16..23}
//   lanes16-31 : K = kc*32 + {8..15} and kc*32 + {24..31}
// -> two contiguous 16-byte loads.
// ---------------------------------------------------------------------------
__device__ __forceinline__ v16h_t load_op16(const _Float16* rowbase, int kc, int lane) {
    const _Float16* p = rowbase + kc * 32 + ((lane >> 4) & 1) * 8;
    v8h_t lo = *(const v8h_t*)(p);
    v8h_t hi = *(const v8h_t*)(p + 16);
    v16h_t r;
#pragma unroll
    for (int t = 0; t < 8; ++t) { r[t] = lo[t]; r[t + 8] = hi[t]; }
    return r;
}

// ---------------------------------------------------------------------------
// Packing kernels
// ---------------------------------------------------------------------------
__global__ __launch_bounds__(256) void pack_wcat_kernel(
    const float* __restrict__ W_ih, const float* __restrict__ W_hh,
    _Float16* __restrict__ Wcat) {
    int id = blockIdx.x * 256 + threadIdx.x;            // 1024*288 threads
    if (id >= GATES * KCAT) return;
    int n = id / KCAT, k = id - n * KCAT;
    float v = (k < OUT_DIM) ? W_ih[n * OUT_DIM + k] : W_hh[n * HID + (k - OUT_DIM)];
    Wcat[id] = (_Float16)v;
}

__global__ __launch_bounds__(256) void pack_bias_kernel(
    const float* __restrict__ b_ih, const float* __restrict__ b_hh,
    float* __restrict__ biascat) {
    int id = blockIdx.x * 256 + threadIdx.x;            // 1024 threads
    if (id < GATES) biascat[id] = b_ih[id] + b_hh[id];
}

__global__ __launch_bounds__(256) void pack_wpool_kernel(
    const float* __restrict__ W_pool, _Float16* __restrict__ Wpool) {
    int id = blockIdx.x * 256 + threadIdx.x;            // 32*256 threads
    if (id < OUT_DIM * HID) Wpool[id] = (_Float16)W_pool[id];
}

__global__ __launch_bounds__(256) void pack_h_kernel(
    const float* __restrict__ h, _Float16* __restrict__ X) {
    int id = blockIdx.x * 256 + threadIdx.x;            // N*HID threads
    if (id >= N_PED * HID) return;
    int i = id >> 8, k = id & 255;
    X[(size_t)i * KCAT + OUT_DIM + k] = (_Float16)h[id];
}

// ---------------------------------------------------------------------------
// Top-4 nearest neighbours + per-neighbour embedding -> X[:, 0:32] (f16)
// obs2 cached in LDS (48 KB). dist^2 preserves top-k order; strict '<'
// insertion keeps earliest index on ties (matches jax.lax.top_k stability).
// ---------------------------------------------------------------------------
__global__ __launch_bounds__(256) void nbr_emb_kernel(
    const float* __restrict__ obs1, const float* __restrict__ obs2,
    const float* __restrict__ W_emb, const float* __restrict__ b_emb,
    _Float16* __restrict__ X) {
    __shared__ float2 spos[N_PED];
    const int tid = threadIdx.x;
    for (int k = tid; k < N_PED; k += 256) spos[k] = ((const float2*)obs2)[k];
    __syncthreads();

    const int i = blockIdx.x * 256 + tid;
    const float2 pi = spos[i];

    float bd0 = INFINITY, bd1 = INFINITY, bd2 = INFINITY, bd3 = INFINITY;
    int   bi0 = 0, bi1 = 0, bi2 = 0, bi3 = 0;
    for (int j = 0; j < N_PED; ++j) {
        float dx = spos[j].x - pi.x;
        float dy = spos[j].y - pi.y;
        float d = dx * dx + dy * dy;
        if (j == i) continue;
        if (d < bd3) {
            bd3 = d; bi3 = j;
            if (bd3 < bd2) { float td = bd2; bd2 = bd3; bd3 = td; int ti = bi2; bi2 = bi3; bi3 = ti; }
            if (bd2 < bd1) { float td = bd1; bd1 = bd2; bd2 = td; int ti = bi1; bi1 = bi2; bi2 = ti; }
            if (bd1 < bd0) { float td = bd0; bd0 = bd1; bd1 = td; int ti = bi0; bi0 = bi1; bi1 = ti; }
        }
    }

    const float vix = pi.x - obs1[2 * i];
    const float viy = pi.y - obs1[2 * i + 1];
    int nb[NTOP] = {bi0, bi1, bi2, bi3};
#pragma unroll
    for (int t = 0; t < NTOP; ++t) {
        int j = nb[t];
        float px = spos[j].x - pi.x;
        float py = spos[j].y - pi.y;
        float vx = (spos[j].x - obs1[2 * j])     - vix;
        float vy = (spos[j].y - obs1[2 * j + 1]) - viy;
#pragma unroll
        for (int e = 0; e < 8; ++e) {
            float v = px * W_emb[e] + py * W_emb[8 + e] +
                      vx * W_emb[16 + e] + vy * W_emb[24 + e] + b_emb[e];
            X[(size_t)i * KCAT + t * 8 + e] = (_Float16)(v > 0.f ? v : 0.f);
        }
    }
}

// ---------------------------------------------------------------------------
// Fused: gates = X @ Wcat^T (WMMA f16->f32, K=288) -> LSTM nonlinearity
//        -> h_new (f16, LDS) -> out = h_new @ Wpool^T + b_pool (WMMA)
// Block = 256 threads (8 waves) owns 32 rows. LDS: 128KB gates + 16KB h.
// ---------------------------------------------------------------------------
__global__ __launch_bounds__(256) void fused_lstm_kernel(
    const _Float16* __restrict__ X, const _Float16* __restrict__ Wcat,
    const float* __restrict__ biascat, const _Float16* __restrict__ Wpool,
    const float* __restrict__ c_in, const float* __restrict__ b_pool,
    float* __restrict__ out) {
    __shared__ float    lds_g[ROWS * GATES];   // 128 KB
    __shared__ _Float16 lds_h[ROWS * HID];     //  16 KB

    const int tid = threadIdx.x;
    const int lane = tid & 31;
    const int w = tid >> 5;
    const int rowbase = blockIdx.x * ROWS;

    // ---- Phase 1: gate GEMM, tiles 16x16, K = 288 (9 chunks of 32) ----
    for (int t = w; t < (ROWS / 16) * (GATES / 16); t += 8) {
        const int rt = t >> 6;                 // 0..1
        const int ct = t & 63;                 // 0..63
        const _Float16* arow = X    + (size_t)(rowbase + rt * 16 + (lane & 15)) * KCAT;
        const _Float16* brow = Wcat + (size_t)(ct * 16 + (lane & 15)) * KCAT;
        v8f_t acc = {};
#pragma unroll
        for (int kc = 0; kc < 9; ++kc) {
            v16h_t a = load_op16(arow, kc, lane);
            v16h_t b = load_op16(brow, kc, lane);
            acc = __builtin_amdgcn_wmma_f32_16x16x32_f16(
                false, a, false, b, (short)0, acc, false, false);
        }
        const int mloc = rt * 16 + ((lane >> 4) & 1) * 8;
        const int nloc = ct * 16 + (lane & 15);
#pragma unroll
        for (int r = 0; r < 8; ++r)
            lds_g[(mloc + r) * GATES + nloc] = acc[r];
    }
    __syncthreads();

    // ---- Phase 2: LSTM elementwise (PyTorch gate order i,f,g,o) ----
    for (int idx = tid; idx < ROWS * HID; idx += 256) {
        const int r = idx >> 8, hcol = idx & 255;
        const int row = rowbase + r;
        float gi = lds_g[r * GATES + hcol]             + biascat[hcol];
        float gf = lds_g[r * GATES + HID + hcol]       + biascat[HID + hcol];
        float gg = lds_g[r * GATES + 2 * HID + hcol]   + biascat[2 * HID + hcol];
        float go = lds_g[r * GATES + 3 * HID + hcol]   + biascat[3 * HID + hcol];
        float co = c_in[(size_t)row * HID + hcol];
        float si = 1.f / (1.f + __expf(-gi));
        float sf = 1.f / (1.f + __expf(-gf));
        float so = 1.f / (1.f + __expf(-go));
        float cn = sf * co + si * tanhf(gg);
        float hn = so * tanhf(cn);
        lds_h[r * HID + hcol] = (_Float16)hn;
    }
    __syncthreads();

    // ---- Phase 3: pooling GEMM 32x32 = lds_h[32,256] @ Wpool^T ----
    if (w < 4) {
        const int rt = w >> 1, ct = w & 1;
        const _Float16* arow = lds_h + (rt * 16 + (lane & 15)) * HID;
        const _Float16* brow = Wpool + (ct * 16 + (lane & 15)) * HID;
        v8f_t acc = {};
#pragma unroll
        for (int kc = 0; kc < 8; ++kc) {
            v16h_t a = load_op16(arow, kc, lane);
            v16h_t b = load_op16(brow, kc, lane);
            acc = __builtin_amdgcn_wmma_f32_16x16x32_f16(
                false, a, false, b, (short)0, acc, false, false);
        }
        const int sel = (lane >> 4) & 1;
        const int colg = ct * 16 + (lane & 15);
        const float bp = b_pool[colg];
#pragma unroll
        for (int r = 0; r < 8; ++r) {
            const int rowg = rowbase + rt * 16 + sel * 8 + r;
            out[(size_t)rowg * OUT_DIM + colg] = acc[r] + bp;
        }
    }
}

// ---------------------------------------------------------------------------
// Launch
// ---------------------------------------------------------------------------
extern "C" void kernel_launch(void* const* d_in, const int* in_sizes, int n_in,
                              void* d_out, int out_size, void* d_ws, size_t ws_size,
                              hipStream_t stream) {
    (void)in_sizes; (void)n_in; (void)out_size; (void)ws_size;
    const float* obs1   = (const float*)d_in[0];
    const float* obs2   = (const float*)d_in[1];
    const float* h      = (const float*)d_in[2];
    const float* c      = (const float*)d_in[3];
    const float* W_emb  = (const float*)d_in[4];
    const float* b_emb  = (const float*)d_in[5];
    const float* W_ih   = (const float*)d_in[6];
    const float* b_ih   = (const float*)d_in[7];
    const float* W_hh   = (const float*)d_in[8];
    const float* b_hh   = (const float*)d_in[9];
    const float* W_pool = (const float*)d_in[10];
    const float* b_pool = (const float*)d_in[11];
    float* out = (float*)d_out;

    // Workspace carve (all offsets 256B-aligned)
    char* ws = (char*)d_ws;
    _Float16* X       = (_Float16*)(ws);                       // N*288 f16   = 3,538,944 B
    _Float16* Wcat    = (_Float16*)(ws + 3538944);             // 1024*288 f16=   589,824 B
    float*    biascat = (float*)   (ws + 4128768);             // 1024 f32    =     4,096 B
    _Float16* Wpool   = (_Float16*)(ws + 4132864);             // 32*256 f16  =    16,384 B

    pack_wcat_kernel <<<(GATES * KCAT + 255) / 256, 256, 0, stream>>>(W_ih, W_hh, Wcat);
    pack_bias_kernel <<<(GATES + 255) / 256,        256, 0, stream>>>(b_ih, b_hh, biascat);
    pack_wpool_kernel<<<(OUT_DIM * HID + 255) / 256,256, 0, stream>>>(W_pool, Wpool);
    pack_h_kernel    <<<(N_PED * HID) / 256,        256, 0, stream>>>(h, X);
    nbr_emb_kernel   <<<N_PED / 256,                256, 0, stream>>>(obs1, obs2, W_emb, b_emb, X);
    fused_lstm_kernel<<<N_PED / ROWS,               256, 0, stream>>>(X, Wcat, biascat, Wpool,
                                                                     c, b_pool, out);
}
